// PPOAgentQuantumJumanji_49117245997359
// MI455X (gfx1250) — matrix operand order repaired and checked by reading, not compiled, not run on hardware
//
#include <hip/hip_runtime.h>

#define NQ     10
#define NA     10
#define NL     5
#define DIM    1024
#define NPAIR  45
#define NFEAT  55     // NA + NPAIR
#define KPAD   64
#define BATCH  2048

typedef __attribute__((ext_vector_type(16))) _Float16 v16h;
typedef __attribute__((ext_vector_type(8)))  _Float16 v8h;
typedef __attribute__((ext_vector_type(8)))  float    v8f;

// ---------------------------------------------------------------------------
// Kernel 1: prep. Xp = f16(x) padded to [BATCH][64]; Zt = f16 ±1 Z-table
// [DIM][64] (k<10: single-qubit sign from bit 9-k; 10<=k<55: pair parity).
// ---------------------------------------------------------------------------
__global__ void qppo_prep_kernel(const float* __restrict__ x,
                                 _Float16* __restrict__ Xp,
                                 _Float16* __restrict__ Zt) {
  int tid = blockIdx.x * blockDim.x + threadIdx.x;
  if (tid < BATCH * KPAD) {
    int b = tid >> 6, k = tid & 63;
    float v = (k < NFEAT) ? x[b * NFEAT + k] : 0.0f;
    Xp[tid] = (_Float16)v;
  }
  if (tid < DIM * KPAD) {
    int d = tid >> 6, k = tid & 63;
    float s = 0.0f;
    if (k < NA) {
      s = ((d >> (NQ - 1 - k)) & 1) ? -1.0f : 1.0f;
    } else if (k < NFEAT) {
      int p = k - NA;
      int i = 0, rem = p, cnt = NQ - 1;
      while (rem >= cnt) { rem -= cnt; cnt--; i++; }
      int j = i + 1 + rem;
      int bi = (d >> (NQ - 1 - i)) & 1;
      int bj = (d >> (NQ - 1 - j)) & 1;
      s = (bi ^ bj) ? -1.0f : 1.0f;
    }
    Zt[tid] = (_Float16)s;
  }
}

// ---------------------------------------------------------------------------
// Kernel 2: expo[2048][1024] = Xp @ Zt^T via V_WMMA_F32_16X16X32_F16.
// One wave per 16x16 tile, 2 k-steps (K=64).
// A layout (ISA 16-bit A 16x32): lane L: M=L%16; halves 0..7 = K kb..kb+7,
//   halves 8..15 = K kb+16..kb+23, kb = (L<16 ? 0 : 8).
// B layout: lane L: N=L%16; halves 0..7 = K kb..kb+7,
//   halves 8..15 = K kb+16..kb+23, kb = (L<16 ? 0 : 8) (same as A).
// ---------------------------------------------------------------------------
__global__ void qppo_expo_wmma_kernel(const _Float16* __restrict__ Xp,
                                      const _Float16* __restrict__ Zt,
                                      float* __restrict__ expo) {
  int lane = threadIdx.x & 31;
  int wave = threadIdx.x >> 5;
  int tile = blockIdx.x * 8 + wave;       // 8192 tiles
  int mt = tile >> 6;                     // 0..127
  int nt = tile & 63;                     // 0..63

  int row  = mt * 16 + (lane & 15);
  int dcol = nt * 16 + (lane & 15);
  int ka = (lane < 16) ? 0 : 8;
  int kb = (lane < 16) ? 0 : 8;

  v8f acc = {};
#pragma unroll
  for (int ks = 0; ks < 2; ++ks) {
    int k0 = ks * 32;
    v8h alo = *(const v8h*)(Xp + row * KPAD + k0 + ka);
    v8h ahi = *(const v8h*)(Xp + row * KPAD + k0 + ka + 16);
    v16h a;
#pragma unroll
    for (int i = 0; i < 8; ++i) { a[i] = alo[i]; a[8 + i] = ahi[i]; }
    v8h blo = *(const v8h*)(Zt + dcol * KPAD + k0 + kb);
    v8h bhi = *(const v8h*)(Zt + dcol * KPAD + k0 + kb + 16);
    v16h b;
#pragma unroll
    for (int i = 0; i < 8; ++i) { b[i] = blo[i]; b[8 + i] = bhi[i]; }
    acc = __builtin_amdgcn_wmma_f32_16x16x32_f16(
        /*neg_a=*/false, a, /*neg_b=*/false, b,
        /*c_mod=*/(short)0, acc, /*reuse_a=*/false, /*reuse_b=*/false);
  }

  // C layout: VGPR r, lanes 0-15: M=r; lanes 16-31: M=8+r; N=lane%16
  int mbase = mt * 16 + ((lane < 16) ? 0 : 8);
  int n = nt * 16 + (lane & 15);
#pragma unroll
  for (int r = 0; r < 8; ++r) {
    expo[(mbase + r) * DIM + n] = acc[r];
  }
}

// ---------------------------------------------------------------------------
// Kernel 3: statevector sim. One wave32 per batch row.
// Amplitude index = slot*32 + lane (slot in 0..31 -> register bits 9..5,
// lane -> bits 4..0). Qubit q bit position = 9-q.
// ---------------------------------------------------------------------------
__device__ __forceinline__ void wave_reduce_all(float& v) {
#pragma unroll
  for (int m = 16; m >= 1; m >>= 1) v += __shfl_xor(v, m, 32);
}

__device__ __forceinline__ void simulate_circuit(const float* __restrict__ expo_row,
                                                 const float* __restrict__ iscale,
                                                 const float* __restrict__ wts,
                                                 int lane, float ev[NQ]) {
  float re[32], im[32];
#pragma unroll
  for (int r = 0; r < 32; ++r) { re[r] = 0.03125f; im[r] = 0.0f; }  // 1/sqrt(1024)

  for (int layer = 0; layer < NL; ++layer) {
    float sc = iscale[layer];
    float wl = wts[layer];
    float sr, cr;
    __sincosf(0.5f * wl, &sr, &cr);

    // diagonal phase: state *= exp(-0.5i * sc * expo)
#pragma unroll
    for (int r = 0; r < 32; ++r) {
      float ph = -0.5f * sc * expo_row[r * 32 + lane];
      float sp, cp;
      __sincosf(ph, &sp, &cp);
      float nr = re[r] * cp - im[r] * sp;
      float ni = re[r] * sp + im[r] * cp;
      re[r] = nr; im[r] = ni;
    }

    // RX on qubits 0..4: register-bit (4-q) butterflies
#pragma unroll
    for (int q = 0; q < 5; ++q) {
      int stride = 16 >> q;
#pragma unroll
      for (int r0 = 0; r0 < 32; ++r0) {
        if ((r0 & stride) == 0) {
          int r1 = r0 + stride;
          float a_re = re[r0], a_im = im[r0];
          float b_re = re[r1], b_im = im[r1];
          re[r0] = cr * a_re + sr * b_im;
          im[r0] = cr * a_im - sr * b_re;
          re[r1] = cr * b_re + sr * a_im;
          im[r1] = cr * b_im - sr * a_re;
        }
      }
    }

    // RX on qubits 5..9: lane-bit (9-q) butterflies via shuffles
#pragma unroll
    for (int qq = 0; qq < 5; ++qq) {
      int mask = 16 >> qq;
#pragma unroll
      for (int r = 0; r < 32; ++r) {
        float pre = __shfl_xor(re[r], mask, 32);
        float pim = __shfl_xor(im[r], mask, 32);
        float nre = cr * re[r] + sr * pim;
        float nim = cr * im[r] - sr * pre;
        re[r] = nre; im[r] = nim;
      }
    }
  }

  // <Z_q> = sum_idx probs[idx] * (+-1)
  float P = 0.0f;
  float S[5] = {0.f, 0.f, 0.f, 0.f, 0.f};
#pragma unroll
  for (int r = 0; r < 32; ++r) {
    float p = re[r] * re[r] + im[r] * im[r];
    P += p;
#pragma unroll
    for (int q = 0; q < 5; ++q) {
      if ((r >> (4 - q)) & 1) S[q] -= p; else S[q] += p;
    }
  }
#pragma unroll
  for (int q = 0; q < 5; ++q) {
    float v = S[q];
    wave_reduce_all(v);
    ev[q] = v;
  }
#pragma unroll
  for (int q = 5; q < 10; ++q) {
    float v = ((lane >> (9 - q)) & 1) ? -P : P;
    wave_reduce_all(v);
    ev[q] = v;
  }
}

__global__ void qppo_sim_kernel(const float* __restrict__ expo,
                                const float* __restrict__ isa,
                                const float* __restrict__ wa,
                                const float* __restrict__ osa,
                                const float* __restrict__ isc,
                                const float* __restrict__ wc,
                                const float* __restrict__ osc,
                                const int*   __restrict__ action,
                                float* __restrict__ out) {
  int lane = threadIdx.x & 31;
  int wave = threadIdx.x >> 5;
  int b = blockIdx.x * 8 + wave;
  const float* erow = expo + b * DIM;

  float eva[NQ];
  simulate_circuit(erow, isa, wa, lane, eva);
  float evc[NQ];
  simulate_circuit(erow, isc, wc, lane, evc);

  // actor head: log-softmax, log_prob, entropy
  float logits[NA];
#pragma unroll
  for (int a = 0; a < NA; ++a) logits[a] = eva[a] * osa[a];
  float m = logits[0];
#pragma unroll
  for (int a = 1; a < NA; ++a) m = fmaxf(m, logits[a]);
  float se = 0.0f;
#pragma unroll
  for (int a = 0; a < NA; ++a) se += __expf(logits[a] - m);
  float lse = m + __logf(se);
  int act = action[b];
  float log_prob = logits[act] - lse;
  float ent = 0.0f;
#pragma unroll
  for (int a = 0; a < NA; ++a) {
    float lp = logits[a] - lse;
    ent -= __expf(lp) * lp;
  }
  float value = evc[0] * osc[0];

  if (lane == 0) {
    out[b * 3 + 0] = log_prob;
    out[b * 3 + 1] = ent;
    out[b * 3 + 2] = value;
  }
}

// ---------------------------------------------------------------------------
extern "C" void kernel_launch(void* const* d_in, const int* in_sizes, int n_in,
                              void* d_out, int out_size, void* d_ws, size_t ws_size,
                              hipStream_t stream) {
  (void)in_sizes; (void)n_in; (void)out_size; (void)ws_size;
  const float* x   = (const float*)d_in[0];
  const float* isa = (const float*)d_in[1];
  const float* wa  = (const float*)d_in[2];
  const float* osa = (const float*)d_in[3];
  const float* isc = (const float*)d_in[4];
  const float* wc  = (const float*)d_in[5];
  const float* osc = (const float*)d_in[6];
  const int*   act = (const int*)d_in[7];
  float* out = (float*)d_out;

  char* ws = (char*)d_ws;
  _Float16* Xp  = (_Float16*)ws;                                   // 256 KB
  _Float16* Zt  = (_Float16*)(ws + (size_t)BATCH * KPAD * 2);      // 128 KB
  float*   expo = (float*)(ws + (size_t)BATCH * KPAD * 2
                              + (size_t)DIM * KPAD * 2);           // 8 MB

  qppo_prep_kernel<<<(BATCH * KPAD + 255) / 256, 256, 0, stream>>>(x, Xp, Zt);
  qppo_expo_wmma_kernel<<<(128 * 64) / 8, 256, 0, stream>>>(Xp, Zt, expo);
  qppo_sim_kernel<<<BATCH / 8, 256, 0, stream>>>(expo, isa, wa, osa, isc, wc, osc, act, out);
}
